// DSKR_36240934044097
// MI455X (gfx1250) — compile-verified, matrix operand order checked by reading, not run on hardware
//
#include <hip/hip_runtime.h>
#include <hip/hip_bf16.h>
#include <math.h>

#define DEV __device__ __forceinline__

typedef __attribute__((ext_vector_type(16))) _Float16 v16h;
typedef __attribute__((ext_vector_type(8)))  float    v8f;

namespace {

constexpr int B_  = 4;
constexpr int NC  = 4096;
constexpr int NT  = 1024;
constexpr int NN  = NC + NT;   // 5120 nodes per batch
constexpr int KK  = 10;        // top-k
constexpr int BNN = B_ * NN;   // 20480 total nodes
constexpr int BNT = B_ * NT;   // 4096 test nodes

// ---- CDNA5 WMMA fragment maps (ISA 7.12.2, wave32) ----
// A (16x32 f16): lane holds row m=lane&15; element e -> K offset:
DEV int a_kidx(int e, int hi) { return (e & 7) + ((e >> 3) << 4) + (hi << 3); }

DEV v8f wmma_f16(v16h a, v16h b, v8f c) {
  return __builtin_amdgcn_wmma_f32_16x16x32_f16(false, a, false, b, (short)0, c,
                                                false, false);
}

// Branch-free tanh-approx gelu: v_exp_f32 + v_rcp_f32, no libcall slow path.
DEV float gelu_t(float x) {
  float y = 0.7978845608028654f * (x + 0.044715f * x * x * x);
  float t = __expf(-2.0f * fabsf(y));
  float th = (1.0f - t) * __builtin_amdgcn_rcpf(1.0f + t);
  th = copysignf(th, y);
  return 0.5f * x * (1.0f + th);
}

// Pre-swizzled weight fragment: one contiguous 32B load per lane.
DEV v16h ldw(const v16h* __restrict__ W, int kt, int nt, int NTt, int lane) {
  return W[(kt * NTt + nt) * 32 + lane];
}

// Scatter-store pointer for writing a C-tile column into an A-fragment-layout
// LDS activation buffer (stride 16 halves per j).
DEV _Float16* act_ptr(_Float16* buf, int n, int hi) {
  return buf + (((n >> 5) * 32 + (((n >> 3) & 1) << 4) + (hi << 3)) << 4)
             + (n & 7) + (((n >> 4) & 1) << 3);
}

// A fragment from fragment-layout LDS buffer: single 32B read.
DEV v16h act_frag(const v16h* __restrict__ buf, int kc, int lane) {
  return buf[kc * 32 + lane];
}

// Build an A fragment from a row-major f32 feature row: the K pattern per lane
// is two runs of 8 consecutive floats -> 4 aligned float4 loads.
DEV v16h pack_row(const float* __restrict__ row, int k0, int hi) {
  const float* base = row + k0 + hi * 8;
  float4 q0 = *(const float4*)(base);
  float4 q1 = *(const float4*)(base + 4);
  float4 q2 = *(const float4*)(base + 16);
  float4 q3 = *(const float4*)(base + 20);
  v16h r;
  r[0] = (_Float16)q0.x; r[1] = (_Float16)q0.y; r[2] = (_Float16)q0.z; r[3] = (_Float16)q0.w;
  r[4] = (_Float16)q1.x; r[5] = (_Float16)q1.y; r[6] = (_Float16)q1.z; r[7] = (_Float16)q1.w;
  r[8]  = (_Float16)q2.x; r[9]  = (_Float16)q2.y; r[10] = (_Float16)q2.z; r[11] = (_Float16)q2.w;
  r[12] = (_Float16)q3.x; r[13] = (_Float16)q3.y; r[14] = (_Float16)q3.z; r[15] = (_Float16)q3.w;
  return r;
}

} // namespace

// -------- weight f32 -> f16 fragment-layout convert (K zero-padded) --------
// dst layout: [batch][kt][nt][lane][16 halves] ; one thread emits one 32B frag.
__global__ void cvt_frag_kernel(const float* __restrict__ src, _Float16* __restrict__ dst,
                                int batch, int Kin, int KT, int Nn) {
  int NTt = Nn >> 4;
  long total = (long)batch * KT * NTt * 32;
  for (long i = (long)blockIdx.x * blockDim.x + threadIdx.x; i < total;
       i += (long)gridDim.x * blockDim.x) {
    int lane = (int)(i & 31);
    long r = i >> 5;
    int nt = (int)(r % NTt); r /= NTt;
    int kt = (int)(r % KT);
    int bb = (int)(r / KT);
    int hi = lane >> 4;
    int n = nt * 16 + (lane & 15);
    v16h v;
#pragma unroll
    for (int e = 0; e < 16; ++e) {
      int k = kt * 32 + a_kidx(e, hi);
      v[e] = (k < Kin) ? (_Float16)src[((long)bb * Kin + k) * Nn + n] : (_Float16)0.f;
    }
    ((v16h*)dst)[i] = v;
  }
}

__global__ void zero_kernel(float* __restrict__ p, int n) {
  int i = blockIdx.x * blockDim.x + threadIdx.x;
  if (i < n) p[i] = 0.f;
}

// ---------------- top-K nearest senders + softmax weights ----------------
// valid_len >= 2048 >> K, so masked senders (d ~ 1.7e6) can never reach the
// top-10: scan only j < valid_len.
__global__ void topk_kernel(const float* __restrict__ s_ctx, const float* __restrict__ s_test,
                            const int* __restrict__ vlen, int* __restrict__ senders,
                            float* __restrict__ dists, float* __restrict__ wts) {
  int row = blockIdx.x * blockDim.x + threadIdx.x;
  if (row >= BNN) return;
  int b = row / NN, n = row % NN;
  const float* rp = (n < NC) ? s_ctx + ((size_t)b * NC + n) * 3
                             : s_test + ((size_t)b * NT + (n - NC)) * 3;
  float rx = rp[0], ry = rp[1], rz = rp[2];
  int vl = vlen[b];
  float bd[KK]; int bi[KK];
#pragma unroll
  for (int p = 0; p < KK; ++p) { bd[p] = 3.0e38f; bi[p] = 0; }
  const float* cb = s_ctx + (size_t)b * NC * 3;
  for (int j = 0; j < vl; ++j) {
    float dx = rx - cb[j * 3], dy = ry - cb[j * 3 + 1], dz = rz - cb[j * 3 + 2];
    float d2 = dx * dx + dy * dy + dz * dz;
    if (d2 < bd[KK - 1]) {               // sorted insert; ties keep earlier index
      bd[KK - 1] = d2; bi[KK - 1] = j;
#pragma unroll
      for (int p = KK - 1; p > 0; --p) {
        if (bd[p] < bd[p - 1]) {
          float tf = bd[p]; bd[p] = bd[p - 1]; bd[p - 1] = tf;
          int   ti = bi[p]; bi[p] = bi[p - 1]; bi[p - 1] = ti;
        }
      }
    }
  }
  float dk[KK], ex[KK], s = 0.f;
#pragma unroll
  for (int p = 0; p < KK; ++p) dk[p] = sqrtf(bd[p]);
  float mx = dk[0];
#pragma unroll
  for (int p = 0; p < KK; ++p) { ex[p] = __expf(mx - dk[p]); s += ex[p]; }
  float inv = 1.f / s;
  size_t base = (size_t)row * KK;
#pragma unroll
  for (int p = 0; p < KK; ++p) {
    senders[base + p] = bi[p]; dists[base + p] = dk[p]; wts[base + p] = ex[p] * inv;
  }
}

// ---------------- embedding MLP (11->256->128->64) + LN : 1 wave = 16 nodes ----------------
__global__ void embed_kernel(const float* __restrict__ s_ctx, const float* __restrict__ f_ctx,
                             const float* __restrict__ s_test, const float* __restrict__ emb_obs,
                             const v16h* __restrict__ W0, const float* __restrict__ b0,
                             const v16h* __restrict__ W1, const float* __restrict__ b1,
                             const v16h* __restrict__ W2, const float* __restrict__ b2,
                             const float* __restrict__ lng, const float* __restrict__ lnb,
                             float* __restrict__ nodes) {
  __shared__ v16h  hA[2][8 * 32];      // 16x256 activations, fragment layout
  __shared__ v16h  hB[2][4 * 32];      // 16x128 activations, fragment layout
  __shared__ float fO[2][16 * 64];     // pre-LN output, row-major
  int lane = threadIdx.x & 31, wv = threadIdx.x >> 5;
  int node0 = (blockIdx.x * 2 + wv) * 16;
  int m = lane & 15, hi = lane >> 4;
  int gg = node0 + m;
  int b = gg / NN, n = gg % NN;
  bool ctx = n < NC;
  // Padded 16x32 input fragment (D_IN = 11, rest zero)
  v16h a0;
#pragma unroll
  for (int e = 0; e < 16; ++e) {
    int k = a_kidx(e, hi);
    float v = 0.f;
    if (k < 4)       v = emb_obs[(ctx ? 1 : 0) * 4 + k];
    else if (k < 7)  v = ctx ? s_ctx[((size_t)b * NC + n) * 3 + (k - 4)]
                             : s_test[((size_t)b * NT + (n - NC)) * 3 + (k - 4)];
    else if (k < 11) v = ctx ? f_ctx[((size_t)b * NC + n) * 4 + (k - 7)] : 0.f;
    a0[e] = (_Float16)v;
  }
  // layer0: 32 -> 256
  for (int nt = 0; nt < 16; ++nt) {
    v8f c = {};
    c = wmma_f16(a0, ldw(W0, 0, nt, 16, lane), c);
    int col = nt * 16 + m; float bv = b0[col];
    _Float16* p = act_ptr((_Float16*)hA[wv], col, hi);
#pragma unroll
    for (int j = 0; j < 8; ++j) p[j * 16] = (_Float16)gelu_t(c[j] + bv);
  }
  __syncthreads();
  // layer1: 256 -> 128
  for (int nt = 0; nt < 8; ++nt) {
    v8f c = {};
    for (int kc = 0; kc < 8; ++kc)
      c = wmma_f16(act_frag(hA[wv], kc, lane), ldw(W1, kc, nt, 8, lane), c);
    int col = nt * 16 + m; float bv = b1[col];
    _Float16* p = act_ptr((_Float16*)hB[wv], col, hi);
#pragma unroll
    for (int j = 0; j < 8; ++j) p[j * 16] = (_Float16)gelu_t(c[j] + bv);
  }
  __syncthreads();
  // layer2: 128 -> 64 (no activation)
  for (int nt = 0; nt < 4; ++nt) {
    v8f c = {};
    for (int kc = 0; kc < 4; ++kc)
      c = wmma_f16(act_frag(hB[wv], kc, lane), ldw(W2, kc, nt, 4, lane), c);
    int col = nt * 16 + m; float bv = b2[col];
#pragma unroll
    for (int j = 0; j < 8; ++j) fO[wv][(j + 8 * hi) * 64 + col] = c[j] + bv;
  }
  __syncthreads();
  if (lane < 16) {
    const float* xr = fO[wv] + m * 64;
    float mean = 0.f, msq = 0.f;
#pragma unroll
    for (int d = 0; d < 64; ++d) { float x = xr[d]; mean += x; msq += x * x; }
    mean *= (1.f / 64.f); msq *= (1.f / 64.f);
    float inv = rsqrtf(msq - mean * mean + 1e-6f);
    float* op = nodes + (size_t)gg * 64;
    for (int d = 0; d < 64; ++d)
      op[d] = (xr[d] - mean) * inv * lng[d] + lnb[d];
  }
}

// ---------------- message MLP + weighted aggregation : 1 wave = 1 node ----------------
__global__ void msg_kernel(const float* __restrict__ nodes, const int* __restrict__ senders,
                           const float* __restrict__ dists, const float* __restrict__ wts,
                           const float* __restrict__ zrow,
                           const v16h* __restrict__ Wm0, const float* __restrict__ bm0,
                           const v16h* __restrict__ Wm1, const float* __restrict__ bm1,
                           float* __restrict__ agg) {
  __shared__ v16h hb[2][4 * 32];       // 16x128 activations, fragment layout
  int lane = threadIdx.x & 31, wv = threadIdx.x >> 5;
  int gg = blockIdx.x * 2 + wv;        // node index
  int b = gg / NN;
  int m = lane & 15, hi = lane >> 4;
  bool vr = m < KK;
  size_t kbase = (size_t)gg * KK;
  int   sid = vr ? senders[kbase + m] : 0;
  float dd  = vr ? dists[kbase + m]   : 0.f;
  const float* srow = vr ? nodes + ((size_t)b * NN + sid) * 64 : zrow;
  v16h af0 = pack_row(srow, 0, hi);
  v16h af1 = pack_row(srow, 32, hi);
  v16h af2;                            // K 64..95: only k==64 (dist) nonzero
#pragma unroll
  for (int e = 0; e < 16; ++e) af2[e] = (_Float16)0.f;
  af2[0] = (_Float16)((hi == 0) ? dd : 0.f);
  for (int nt = 0; nt < 8; ++nt) {     // 96 -> 128
    v8f c = {};
    c = wmma_f16(af0, ldw(Wm0, 0, nt, 8, lane), c);
    c = wmma_f16(af1, ldw(Wm0, 1, nt, 8, lane), c);
    c = wmma_f16(af2, ldw(Wm0, 2, nt, 8, lane), c);
    int col = nt * 16 + m; float bv = bm0[col];
    _Float16* p = act_ptr((_Float16*)hb[wv], col, hi);
#pragma unroll
    for (int j = 0; j < 8; ++j) p[j * 16] = (_Float16)gelu_t(c[j] + bv);
  }
  __syncthreads();
  float wrow[8];
#pragma unroll
  for (int j = 0; j < 8; ++j) {
    int r = j + 8 * hi;
    wrow[j] = (r < KK) ? wts[kbase + r] : 0.f;
  }
  for (int nt = 0; nt < 4; ++nt) {     // 128 -> 64, then weighted row-sum
    v8f c = {};
    for (int kc = 0; kc < 4; ++kc)
      c = wmma_f16(act_frag(hb[wv], kc, lane), ldw(Wm1, kc, nt, 4, lane), c);
    int col = nt * 16 + m; float bv = bm1[col];
    float part = 0.f;
#pragma unroll
    for (int j = 0; j < 8; ++j) part += wrow[j] * (c[j] + bv);
    part += __shfl_xor(part, 16, 32);  // combine row halves (cols match across halves)
    if (lane < 16) agg[(size_t)gg * 64 + col] = part;
  }
}

// ---------------- node update MLP + residual LayerNorm (in-place) : 1 wave = 16 nodes ----------------
__global__ void upd_kernel(float* __restrict__ nodes, const float* __restrict__ agg,
                           const v16h* __restrict__ Wu0, const float* __restrict__ bu0,
                           const v16h* __restrict__ Wu1, const float* __restrict__ bu1,
                           const float* __restrict__ lng, const float* __restrict__ lnb) {
  __shared__ v16h  hb[2][4 * 32];      // 16x128 activations, fragment layout
  __shared__ float ub[2][16 * 64];     // update, row-major
  int lane = threadIdx.x & 31, wv = threadIdx.x >> 5;
  int node0 = (blockIdx.x * 2 + wv) * 16;
  int m = lane & 15, hi = lane >> 4;
  int gg = node0 + m;
  const float* xrow = nodes + (size_t)gg * 64;
  const float* arow = agg + (size_t)gg * 64;
  v16h af[4];                          // u_in = [nodes(64), agg(64)]
  af[0] = pack_row(xrow, 0, hi);
  af[1] = pack_row(xrow, 32, hi);
  af[2] = pack_row(arow, 0, hi);
  af[3] = pack_row(arow, 32, hi);
  for (int nt = 0; nt < 8; ++nt) {     // 128 -> 128
    v8f c = {};
#pragma unroll
    for (int kc = 0; kc < 4; ++kc)
      c = wmma_f16(af[kc], ldw(Wu0, kc, nt, 8, lane), c);
    int col = nt * 16 + m; float bv = bu0[col];
    _Float16* p = act_ptr((_Float16*)hb[wv], col, hi);
#pragma unroll
    for (int j = 0; j < 8; ++j) p[j * 16] = (_Float16)gelu_t(c[j] + bv);
  }
  __syncthreads();
  for (int nt = 0; nt < 4; ++nt) {     // 128 -> 64
    v8f c = {};
    for (int kc = 0; kc < 4; ++kc)
      c = wmma_f16(act_frag(hb[wv], kc, lane), ldw(Wu1, kc, nt, 4, lane), c);
    int col = nt * 16 + m; float bv = bu1[col];
#pragma unroll
    for (int j = 0; j < 8; ++j) ub[wv][(j + 8 * hi) * 64 + col] = c[j] + bv;
  }
  __syncthreads();
  if (lane < 16) {
    const float* ur = ub[wv] + m * 64;
    float mean = 0.f, msq = 0.f;
#pragma unroll
    for (int d = 0; d < 64; ++d) { float x = xrow[d] + ur[d]; mean += x; msq += x * x; }
    mean *= (1.f / 64.f); msq *= (1.f / 64.f);
    float inv = rsqrtf(msq - mean * mean + 1e-6f);
    float* op = nodes + (size_t)gg * 64;  // read-then-write per element: in-place safe
    for (int d = 0; d < 64; ++d) {
      float x = xrow[d] + ur[d];
      op[d] = (x - mean) * inv * lng[d] + lnb[d];
    }
  }
}

// ---------------- head MLP (64->256->64->2) on test nodes : 1 wave = 16 nodes ----------------
__global__ void head_kernel(const float* __restrict__ nodes,
                            const v16h* __restrict__ W0, const float* __restrict__ b0,
                            const v16h* __restrict__ W1, const float* __restrict__ b1,
                            const float* __restrict__ W2, const float* __restrict__ b2,
                            float* __restrict__ out) {
  __shared__ v16h     h0s[2][8 * 32];  // 16x256 activations, fragment layout
  __shared__ _Float16 h1s[2][16 * 64]; // 16x64, row-major (scalar tail)
  int lane = threadIdx.x & 31, wv = threadIdx.x >> 5;
  int t0 = (blockIdx.x * 2 + wv) * 16;
  int m = lane & 15, hi = lane >> 4;
  int bt = t0 + m;
  int b = bt / NT, t = bt % NT;
  const float* xrow = nodes + ((size_t)b * NN + NC + t) * 64;
  v16h af0 = pack_row(xrow, 0, hi);
  v16h af1 = pack_row(xrow, 32, hi);
  for (int nt = 0; nt < 16; ++nt) {    // 64 -> 256
    v8f c = {};
    c = wmma_f16(af0, ldw(W0, 0, nt, 16, lane), c);
    c = wmma_f16(af1, ldw(W0, 1, nt, 16, lane), c);
    int col = nt * 16 + m; float bv = b0[col];
    _Float16* p = act_ptr((_Float16*)h0s[wv], col, hi);
#pragma unroll
    for (int j = 0; j < 8; ++j) p[j * 16] = (_Float16)gelu_t(c[j] + bv);
  }
  __syncthreads();
  for (int nt = 0; nt < 4; ++nt) {     // 256 -> 64
    v8f c = {};
    for (int kc = 0; kc < 8; ++kc)
      c = wmma_f16(act_frag(h0s[wv], kc, lane), ldw(W1, kc, nt, 4, lane), c);
    int col = nt * 16 + m; float bv = b1[col];
#pragma unroll
    for (int j = 0; j < 8; ++j)
      h1s[wv][(j + 8 * hi) * 64 + col] = (_Float16)gelu_t(c[j] + bv);
  }
  __syncthreads();
  if (lane < 16) {                     // 64 -> 2, then mu / std=exp(lv/2)
    float s0 = b2[0], s1 = b2[1];
#pragma unroll
    for (int j = 0; j < 64; ++j) {
      float h = (float)h1s[wv][m * 64 + j];
      s0 += h * W2[j * 2 + 0];
      s1 += h * W2[j * 2 + 1];
    }
    out[bt] = s0;
    out[BNT + bt] = __expf(0.5f * s1);
  }
}

// ------------------------------------------------------------------------------
extern "C" void kernel_launch(void* const* d_in, const int* in_sizes, int n_in,
                              void* d_out, int out_size, void* d_ws, size_t ws_size,
                              hipStream_t stream) {
  (void)in_sizes; (void)n_in; (void)out_size; (void)ws_size;
  const float* s_ctx   = (const float*)d_in[0];
  const float* f_ctx   = (const float*)d_in[1];
  const float* s_test  = (const float*)d_in[2];
  const int*   vlen    = (const int*)d_in[3];
  const float* emb_obs = (const float*)d_in[4];
  const float* ea_W0   = (const float*)d_in[5];
  const float* ea_b0   = (const float*)d_in[6];
  const float* ea_W1   = (const float*)d_in[7];
  const float* ea_b1   = (const float*)d_in[8];
  const float* ea_W2   = (const float*)d_in[9];
  const float* ea_b2   = (const float*)d_in[10];
  const float* ln_g    = (const float*)d_in[11];
  const float* ln_b    = (const float*)d_in[12];
  const float* blk_Wm0 = (const float*)d_in[13];
  const float* blk_bm0 = (const float*)d_in[14];
  const float* blk_Wm1 = (const float*)d_in[15];
  const float* blk_bm1 = (const float*)d_in[16];
  const float* blk_Wu0 = (const float*)d_in[17];
  const float* blk_bu0 = (const float*)d_in[18];
  const float* blk_Wu1 = (const float*)d_in[19];
  const float* blk_bu1 = (const float*)d_in[20];
  const float* blk_lng = (const float*)d_in[21];
  const float* blk_lnb = (const float*)d_in[22];
  const float* hd_W0   = (const float*)d_in[23];
  const float* hd_b0   = (const float*)d_in[24];
  const float* hd_W1   = (const float*)d_in[25];
  const float* hd_b1   = (const float*)d_in[26];
  const float* hd_W2   = (const float*)d_in[27];
  const float* hd_b2   = (const float*)d_in[28];

  char* ws = (char*)d_ws;
  size_t off = 0;
  auto take = [&](size_t bytes) -> void* {
    off = (off + 255) & ~(size_t)255;
    void* p = ws + off; off += bytes; return p;
  };
  int*      senders = (int*)take((size_t)BNN * KK * 4);
  float*    dists   = (float*)take((size_t)BNN * KK * 4);
  float*    wsoft   = (float*)take((size_t)BNN * KK * 4);
  float*    nodesb  = (float*)take((size_t)BNN * 64 * 4);
  float*    aggb    = (float*)take((size_t)BNN * 64 * 4);
  float*    zrow    = (float*)take(64 * 4);
  // fragment-layout f16 weights: size = batch*KT*NTtiles*32 v16h (32B each)
  _Float16* w_ea0 = (_Float16*)take((size_t)1 * 16 * 32 * 32);
  _Float16* w_ea1 = (_Float16*)take((size_t)8 * 8  * 32 * 32);
  _Float16* w_ea2 = (_Float16*)take((size_t)4 * 4  * 32 * 32);
  _Float16* w_m0  = (_Float16*)take((size_t)6 * 3 * 8 * 32 * 32);
  _Float16* w_m1  = (_Float16*)take((size_t)6 * 4 * 4 * 32 * 32);
  _Float16* w_u0  = (_Float16*)take((size_t)6 * 4 * 8 * 32 * 32);
  _Float16* w_u1  = (_Float16*)take((size_t)6 * 4 * 4 * 32 * 32);
  _Float16* w_h0  = (_Float16*)take((size_t)2 * 16 * 32 * 32);
  _Float16* w_h1  = (_Float16*)take((size_t)8 * 4  * 32 * 32);

  auto cvt = [&](const float* s, _Float16* d, int batch, int Kin, int KT, int Nn) {
    long total = (long)batch * KT * (Nn >> 4) * 32;
    int grid = (int)((total + 255) / 256);
    cvt_frag_kernel<<<grid, 256, 0, stream>>>(s, d, batch, Kin, KT, Nn);
  };
  cvt(ea_W0,   w_ea0, 1,  11, 1, 256);
  cvt(ea_W1,   w_ea1, 1, 256, 8, 128);
  cvt(ea_W2,   w_ea2, 1, 128, 4,  64);
  cvt(blk_Wm0, w_m0,  6,  65, 3, 128);
  cvt(blk_Wm1, w_m1,  6, 128, 4,  64);
  cvt(blk_Wu0, w_u0,  6, 128, 4, 128);
  cvt(blk_Wu1, w_u1,  6, 128, 4,  64);
  cvt(hd_W0,   w_h0,  1,  64, 2, 256);
  cvt(hd_W1,   w_h1,  1, 256, 8,  64);
  zero_kernel<<<1, 64, 0, stream>>>(zrow, 64);

  topk_kernel<<<(BNN + 63) / 64, 64, 0, stream>>>(s_ctx, s_test, vlen,
                                                  senders, dists, wsoft);

  embed_kernel<<<BNN / 32, 64, 0, stream>>>(s_ctx, f_ctx, s_test, emb_obs,
                                            (const v16h*)w_ea0, ea_b0,
                                            (const v16h*)w_ea1, ea_b1,
                                            (const v16h*)w_ea2, ea_b2,
                                            ln_g, ln_b, nodesb);

  for (int i = 0; i < 6; ++i) {
    msg_kernel<<<BNN / 2, 64, 0, stream>>>(nodesb, senders, dists, wsoft, zrow,
                                           (const v16h*)(w_m0 + (size_t)i * 3 * 8 * 32 * 16),
                                           blk_bm0 + (size_t)i * 128,
                                           (const v16h*)(w_m1 + (size_t)i * 4 * 4 * 32 * 16),
                                           blk_bm1 + (size_t)i * 64,
                                           aggb);
    upd_kernel<<<BNN / 32, 64, 0, stream>>>(nodesb, aggb,
                                            (const v16h*)(w_u0 + (size_t)i * 4 * 8 * 32 * 16),
                                            blk_bu0 + (size_t)i * 128,
                                            (const v16h*)(w_u1 + (size_t)i * 4 * 4 * 32 * 16),
                                            blk_bu1 + (size_t)i * 64,
                                            blk_lng + (size_t)i * 64, blk_lnb + (size_t)i * 64);
  }

  head_kernel<<<BNT / 32, 64, 0, stream>>>(nodesb,
                                           (const v16h*)w_h0, hd_b0,
                                           (const v16h*)w_h1, hd_b1,
                                           hd_W2, hd_b2, (float*)d_out);
}
